// Transformer_27874337751621
// MI455X (gfx1250) — matrix operand order repaired and checked by reading, not compiled, not run on hardware
//
#include <hip/hip_runtime.h>

// ---------------------------------------------------------------------------
// Transformer encoder-decoder forward for MI455X (gfx1250, wave32, WMMA).
// All GEMMs go through v_wmma_f32_16x16x32_bf16 (fp32 -> bf16 inputs, fp32
// accumulate).  One generic batched/strided GEMM kernel covers QKV / scores
// (NT) / AV / O-proj / FF / tied head.  Staging is wide & branchless:
// float4 global loads -> v_cvt_pk_bf16_f32 -> ds_store_b128.
// ---------------------------------------------------------------------------

#define D_MODEL 1024
#define NHEAD   16
#define DKH     64
#define DFF_    4096
#define NSEQ    512
#define NB      4
#define MTOK    2048   // NB * NSEQ
#define VOCAB   32000

typedef __attribute__((ext_vector_type(16))) __bf16 v16bf;
typedef __attribute__((ext_vector_type(2)))  __bf16 v2bf;
typedef __attribute__((ext_vector_type(8)))  float  v8f;

union ABfrag { unsigned int u[8]; v16bf v; };

// Pack two f32 into one dword of bf16 (RNE).  Prefer the native packed
// convert; otherwise let clang lower fptrunc to the target's bf16 cvt.
static __device__ __forceinline__ unsigned int pack_bf2(float lo, float hi) {
#if __has_builtin(__builtin_amdgcn_cvt_pk_bf16_f32)
  union { v2bf v; unsigned int u; } cv;
  cv.v = __builtin_amdgcn_cvt_pk_bf16_f32(lo, hi);
  return cv.u;
#else
  union { v2bf v; unsigned int u; } cv;
  cv.v[0] = (__bf16)lo;
  cv.v[1] = (__bf16)hi;
  return cv.u;
#endif
}

// ---------------------------------------------------------------------------
// Generic batched WMMA GEMM:
//   C[z][m][n] = act( alpha * sum_k A[z][m][k] * W[z](k,n) + bias[h][n] ) + add
// z = blockIdx.z, split as b = z / Hdiv, h = z % Hdiv.
// A:  A + b*sAb + h*sAh, element (m,k) at m*lda + k            (row major)
// W:  W + b*sWb + h*sWh, element (k,n) at k*sWk + n*sWn        (NN or NT)
// C:  C + b*sCb + h*sCh, element (m,n) at m*ldc + n
// Contract (all call sites obey): M % 128 == 0, K % 32 == 0, all A/W row
// bases 16-float aligned.  Only N may be ragged.
// ---------------------------------------------------------------------------
struct GemmP {
  const float* A; const float* W; const float* bias; const float* add; float* C;
  int M, N, K, lda, ldc;
  long long sAb, sAh, sWb, sWh, sCb, sCh;
  int sWk, sWn, biasH, relu, Hdiv;
  float alpha;
};

#define TM 128
#define TN 128
#define TK 32
#define LDSP 40   // padded LDS row stride (bf16 elems); 80 B = 16-B aligned

__global__ __launch_bounds__(256) void wmma_gemm(GemmP p) {
  __shared__ unsigned short As[TM * LDSP];
  __shared__ unsigned short Bs[TN * LDSP];

  const int tid  = threadIdx.x;
  const int lane = tid & 31;
  const int w    = tid >> 5;       // 8 waves
  const int wm   = w >> 1;         // 0..3 -> 32-row strip
  const int wn   = w & 1;          // 0..1 -> 64-col strip
  const int hl   = lane >> 4;      // lane half
  const int l15  = lane & 15;

  const int m0 = blockIdx.y * TM;
  const int n0 = blockIdx.x * TN;
  const int z  = blockIdx.z;
  const int bb = z / p.Hdiv;
  const int hh = z % p.Hdiv;

  const float* Ap = p.A + (long long)bb * p.sAb + (long long)hh * p.sAh;
  const float* Wp = p.W + (long long)bb * p.sWb + (long long)hh * p.sWh;
  float*       Cp = p.C + (long long)bb * p.sCb + (long long)hh * p.sCh;
  const float* Ad = p.add ? p.add + (long long)bb * p.sCb + (long long)hh * p.sCh : nullptr;
  const float* Bi = p.bias ? p.bias + (long long)hh * p.biasH : nullptr;

  // staging coordinates (2 threads per row, 16 contiguous k each)
  const int sr  = tid >> 1;
  const int sc0 = (tid & 1) * 16;
  const int gnB  = n0 + sr;
  const bool okB = gnB < p.N;
  const int gnBc = okB ? gnB : (p.N - 1);          // clamped: loads always legal
  const unsigned int bmsk = okB ? 0xFFFFFFFFu : 0u;

  v8f acc[2][4];
  const v8f zf = {0.f, 0.f, 0.f, 0.f, 0.f, 0.f, 0.f, 0.f};
#pragma unroll
  for (int i = 0; i < 2; ++i)
#pragma unroll
    for (int j = 0; j < 4; ++j) acc[i][j] = zf;

  for (int k0 = 0; k0 < p.K; k0 += TK) {
    __syncthreads();
    // ---- stage A tile (TM x TK): unconditional wide loads ----
    {
      const float4* ap =
          (const float4*)(Ap + (long long)(m0 + sr) * p.lda + (k0 + sc0));
      const float4 a0 = ap[0], a1 = ap[1], a2 = ap[2], a3 = ap[3];
      uint4 s0, s1;
      s0.x = pack_bf2(a0.x, a0.y); s0.y = pack_bf2(a0.z, a0.w);
      s0.z = pack_bf2(a1.x, a1.y); s0.w = pack_bf2(a1.z, a1.w);
      s1.x = pack_bf2(a2.x, a2.y); s1.y = pack_bf2(a2.z, a2.w);
      s1.z = pack_bf2(a3.x, a3.y); s1.w = pack_bf2(a3.z, a3.w);
      uint4* dst = (uint4*)&As[sr * LDSP + sc0];
      dst[0] = s0; dst[1] = s1;
    }
    // ---- stage B tile as Bs[n][k]: clamped loads, masked to zero after ----
    {
      float fv[16];
      if (p.sWk == 1) {   // NT path (scores, tied head): contiguous in k
        const float4* wp =
            (const float4*)(Wp + (long long)gnBc * p.sWn + (k0 + sc0));
        const float4 b0 = wp[0], b1 = wp[1], b2 = wp[2], b3 = wp[3];
        fv[0]=b0.x; fv[1]=b0.y; fv[2]=b0.z; fv[3]=b0.w;
        fv[4]=b1.x; fv[5]=b1.y; fv[6]=b1.z; fv[7]=b1.w;
        fv[8]=b2.x; fv[9]=b2.y; fv[10]=b2.z; fv[11]=b2.w;
        fv[12]=b3.x; fv[13]=b3.y; fv[14]=b3.z; fv[15]=b3.w;
      } else {            // NN path: strided weight transpose read
        const float* wp = Wp + (long long)(k0 + sc0) * p.sWk + (long long)gnBc * p.sWn;
#pragma unroll
        for (int i = 0; i < 16; ++i) fv[i] = wp[(long long)i * p.sWk];
      }
      uint4 s0, s1;
      s0.x = pack_bf2(fv[0],  fv[1])  & bmsk;  s0.y = pack_bf2(fv[2],  fv[3])  & bmsk;
      s0.z = pack_bf2(fv[4],  fv[5])  & bmsk;  s0.w = pack_bf2(fv[6],  fv[7])  & bmsk;
      s1.x = pack_bf2(fv[8],  fv[9])  & bmsk;  s1.y = pack_bf2(fv[10], fv[11]) & bmsk;
      s1.z = pack_bf2(fv[12], fv[13]) & bmsk;  s1.w = pack_bf2(fv[14], fv[15]) & bmsk;
      uint4* dst = (uint4*)&Bs[sr * LDSP + sc0];
      dst[0] = s0; dst[1] = s1;
    }
    __syncthreads();

    // ---- build fragments (A and B both use k = 16*half + s) and issue WMMAs ----
    ABfrag afr[2], bfr[4];
#pragma unroll
    for (int fm = 0; fm < 2; ++fm) {
      const int row = wm * 32 + fm * 16 + l15;      // A matrix row M
      const unsigned short* rp = &As[row * LDSP];
#pragma unroll
      for (int j = 0; j < 8; ++j) {
        // VGPR j holds K pair 2j; +16 for upper half
        const int kk = 2 * j + hl * 16;
        afr[fm].u[j] = *(const unsigned int*)&rp[kk];
      }
    }
#pragma unroll
    for (int fn = 0; fn < 4; ++fn) {
      const int row = wn * 64 + fn * 16 + l15;      // B matrix col N
      const unsigned short* rp = &Bs[row * LDSP + hl * 16];
#pragma unroll
      for (int j = 0; j < 8; ++j) bfr[fn].u[j] = *(const unsigned int*)&rp[2 * j];
    }
#pragma unroll
    for (int fm = 0; fm < 2; ++fm)
#pragma unroll
      for (int fn = 0; fn < 4; ++fn)
        acc[fm][fn] = __builtin_amdgcn_wmma_f32_16x16x32_bf16(
            false, afr[fm].v, false, bfr[fn].v, (short)0, acc[fm][fn], false, false);
  }

  // ---- epilogue: alpha, bias, relu, residual add, store (M tiles full) ----
#pragma unroll
  for (int fm = 0; fm < 2; ++fm) {
#pragma unroll
    for (int fn = 0; fn < 4; ++fn) {
      const int gn  = n0 + wn * 64 + fn * 16 + l15;
      const int gmb = m0 + wm * 32 + fm * 16 + hl * 8;
      if (gn < p.N) {
        const float bv = Bi ? Bi[gn] : 0.0f;
#pragma unroll
        for (int r = 0; r < 8; ++r) {
          float v = acc[fm][fn][r] * p.alpha + bv;
          if (p.relu) v = fmaxf(v, 0.0f);
          const long long ci = (long long)(gmb + r) * p.ldc + gn;
          if (Ad) v += Ad[ci];
          Cp[ci] = v;
        }
      }
    }
  }
}

// ---------------------------------------------------------------------------
// LayerNorm over D=1024, one token per block (256 threads, 4 elems/thread).
// ---------------------------------------------------------------------------
__global__ __launch_bounds__(256) void layernorm_k(const float* __restrict__ in,
                                                   const float* __restrict__ g,
                                                   const float* __restrict__ b,
                                                   float* __restrict__ out) {
  __shared__ float red[256];
  const int t   = blockIdx.x;
  const int tid = threadIdx.x;
  const float* x = in + (long long)t * D_MODEL;
  float v4[4];
  float s = 0.f;
#pragma unroll
  for (int i = 0; i < 4; ++i) { v4[i] = x[tid + i * 256]; s += v4[i]; }
  red[tid] = s; __syncthreads();
  for (int st = 128; st; st >>= 1) { if (tid < st) red[tid] += red[tid + st]; __syncthreads(); }
  const float mean = red[0] * (1.0f / D_MODEL);
  __syncthreads();
  float vs = 0.f;
#pragma unroll
  for (int i = 0; i < 4; ++i) { float d = v4[i] - mean; vs += d * d; }
  red[tid] = vs; __syncthreads();
  for (int st = 128; st; st >>= 1) { if (tid < st) red[tid] += red[tid + st]; __syncthreads(); }
  const float rs = rsqrtf(red[0] * (1.0f / D_MODEL) + 1e-5f);
  float* o = out + (long long)t * D_MODEL;
#pragma unroll
  for (int i = 0; i < 4; ++i) {
    const int d = tid + i * 256;
    o[d] = (v4[i] - mean) * rs * g[d] + b[d];
  }
}

// ---------------------------------------------------------------------------
// Row softmax over 512 cols, one wave32 per row; rows ordered (b, h, q).
// ---------------------------------------------------------------------------
__global__ void softmax_rows(float* __restrict__ S, int causal) {
  const int row  = blockIdx.x;
  const int q    = row % NSEQ;
  const int lane = threadIdx.x;
  float* ptr = S + (long long)row * NSEQ;
  float vals[16];
  float mx = -1e30f;
#pragma unroll
  for (int i = 0; i < 16; ++i) {
    const int c = lane + i * 32;
    float v = ptr[c];
    if (causal && c > q) v = -1e30f;
    vals[i] = v;
    mx = fmaxf(mx, v);
  }
  for (int off = 16; off; off >>= 1) mx = fmaxf(mx, __shfl_xor(mx, off, 32));
  float sum = 0.f;
#pragma unroll
  for (int i = 0; i < 16; ++i) { float e = __expf(vals[i] - mx); vals[i] = e; sum += e; }
  for (int off = 16; off; off >>= 1) sum += __shfl_xor(sum, off, 32);
  const float inv = 1.0f / sum;
#pragma unroll
  for (int i = 0; i < 16; ++i) ptr[lane + i * 32] = vals[i] * inv;
}

// ---------------------------------------------------------------------------
// Embedding gather: out[t] = emb[id[t]] + pos[id[t]]   (faithful source bug:
// positional embedding is indexed by TOKEN ID).
// ---------------------------------------------------------------------------
__global__ __launch_bounds__(256) void embed_k(const int* __restrict__ ids,
                                               const float* __restrict__ emb,
                                               const float* __restrict__ pos,
                                               float* __restrict__ out) {
  const int t  = blockIdx.x;
  const int id = ids[t];
  const float* e  = emb + (long long)id * D_MODEL;
  const float* pp = pos + (long long)id * D_MODEL;
  float* o = out + (long long)t * D_MODEL;
#pragma unroll
  for (int i = 0; i < 4; ++i) {
    const int d = threadIdx.x + i * 256;
    o[d] = e[d] + pp[d];
  }
}

// ---------------------------------------------------------------------------
// Host orchestration.
// Input leaf order assumes jax-pytree flattening (dict keys sorted):
//   0:X 1:Y 2:context
//   3..170: dec layers (28 leaves each):
//     +0..7  cattn: bk,bo,bq,bv,wk,wo,wq,wv
//     +8..11 ff:    b1,b2,w1,w2
//     +12..19 ln0_b,ln0_g,ln1_b,ln1_g,ln2_b,ln2_g,ln3_b,ln3_g
//     +20..27 sattn: bk,bo,bq,bv,wk,wo,wq,wv
//   171:dec_ln_b 172:dec_ln_g 173:dec_pos 174:emb
//   175..270: enc layers (16 leaves each):
//     +0..7 attn, +8..11 ff, +12..15 ln1_b,ln1_g,ln2_b,ln2_g
//   271:enc_ln_b 272:enc_ln_g 273:enc_pos 274:head_b
// ---------------------------------------------------------------------------
extern "C" void kernel_launch(void* const* d_in, const int* in_sizes, int n_in,
                              void* d_out, int out_size, void* d_ws, size_t ws_size,
                              hipStream_t stream) {
  (void)in_sizes; (void)n_in; (void)out_size; (void)ws_size;
  auto P = [&](int i) { return (const float*)d_in[i]; };
  const int* Xids = (const int*)d_in[0];
  const int* Cids = (const int*)d_in[2];

  float* ws    = (float*)d_ws;
  float* cbuf  = ws;                                    // 2M floats
  float* xbuf  = ws + 1 * 2097152;
  float* tbuf  = ws + 2 * 2097152;
  float* t2buf = ws + 3 * 2097152;
  float* qbuf  = ws + 4 * 2097152;
  float* kbuf  = ws + 5 * 2097152;
  float* vbuf  = ws + 6 * 2097152;
  float* obuf  = ws + 7 * 2097152;
  float* ffbuf = ws + 8 * 2097152;                      // 8M floats
  float* sbuf  = ws + 8 * 2097152 + 8388608;            // 16.78M floats

  auto gemm = [&](const float* A, const float* Wt, const float* bias, int biasH,
                  const float* add, float* C,
                  int M, int N, int K, int lda, long long sAb, long long sAh,
                  int sWk, int sWn, long long sWb, long long sWh,
                  int ldc, long long sCb, long long sCh,
                  float alpha, int relu, int Hdiv, int Z) {
    GemmP p;
    p.A = A; p.W = Wt; p.bias = bias; p.add = add; p.C = C;
    p.M = M; p.N = N; p.K = K; p.lda = lda; p.ldc = ldc;
    p.sAb = sAb; p.sAh = sAh; p.sWb = sWb; p.sWh = sWh; p.sCb = sCb; p.sCh = sCh;
    p.sWk = sWk; p.sWn = sWn; p.biasH = biasH; p.relu = relu; p.Hdiv = Hdiv;
    p.alpha = alpha;
    dim3 grid((N + TN - 1) / TN, (M + TM - 1) / TM, Z);
    wmma_gemm<<<grid, 256, 0, stream>>>(p);
  };

  auto lnrm = [&](const float* in, const float* g, const float* b, float* out) {
    layernorm_k<<<MTOK, 256, 0, stream>>>(in, g, b, out);
  };

  // Multi-head attention. ab = leaf index of this attention's 'bk'.
  // Q/K/V live in "concat head" layout: [token][h*64+dk], ld = 1024.
  auto attn = [&](const float* xq, const float* xkv, int ab, int causal,
                  const float* res, float* out) {
    const float* bk = P(ab + 0); const float* bo = P(ab + 1);
    const float* bq = P(ab + 2); const float* bv = P(ab + 3);
    const float* wk = P(ab + 4); const float* wo = P(ab + 5);
    const float* wq = P(ab + 6); const float* wv = P(ab + 7);
    const long long sW = (long long)D_MODEL * DKH;     // per-head weight stride
    // Q/K/V projections, batched over heads (M=2048, N=64, K=1024)
    gemm(xq,  wq, bq, DKH, nullptr, qbuf, MTOK, DKH, D_MODEL, D_MODEL, 0, 0,
         DKH, 1, 0, sW, D_MODEL, 0, DKH, 1.0f, 0, NHEAD, NHEAD);
    gemm(xkv, wk, bk, DKH, nullptr, kbuf, MTOK, DKH, D_MODEL, D_MODEL, 0, 0,
         DKH, 1, 0, sW, D_MODEL, 0, DKH, 1.0f, 0, NHEAD, NHEAD);
    gemm(xkv, wv, bv, DKH, nullptr, vbuf, MTOK, DKH, D_MODEL, D_MODEL, 0, 0,
         DKH, 1, 0, sW, D_MODEL, 0, DKH, 1.0f, 0, NHEAD, NHEAD);
    // scores = (Q @ K^T) / 8, batched over (b,h): M=N=512, K=64  (NT)
    gemm(qbuf, kbuf, nullptr, 0, nullptr, sbuf, NSEQ, NSEQ, DKH,
         D_MODEL, (long long)NSEQ * D_MODEL, DKH,
         1, D_MODEL, (long long)NSEQ * D_MODEL, DKH,
         NSEQ, (long long)NHEAD * NSEQ * NSEQ, (long long)NSEQ * NSEQ,
         0.125f, 0, NHEAD, NB * NHEAD);
    softmax_rows<<<NB * NHEAD * NSEQ, 32, 0, stream>>>(sbuf, causal);
    // O_h = P @ V, batched over (b,h): M=512, N=64, K=512  (NN)
    gemm(sbuf, vbuf, nullptr, 0, nullptr, obuf, NSEQ, DKH, NSEQ,
         NSEQ, (long long)NHEAD * NSEQ * NSEQ, (long long)NSEQ * NSEQ,
         D_MODEL, 1, (long long)NSEQ * D_MODEL, DKH,
         D_MODEL, (long long)NSEQ * D_MODEL, DKH, 1.0f, 0, NHEAD, NB * NHEAD);
    // out = res + concat(O) @ wo + bo
    gemm(obuf, wo, bo, 0, res, out, MTOK, D_MODEL, D_MODEL, D_MODEL, 0, 0,
         D_MODEL, 1, 0, 0, D_MODEL, 0, 0, 1.0f, 0, 1, 1);
  };

  // Feed-forward. fb = leaf index of 'b1'.
  auto ffn = [&](const float* xin, int fb, const float* res, float* out) {
    gemm(xin, P(fb + 2), P(fb + 0), 0, nullptr, ffbuf, MTOK, DFF_, D_MODEL,
         D_MODEL, 0, 0, DFF_, 1, 0, 0, DFF_, 0, 0, 1.0f, 1, 1, 1);
    gemm(ffbuf, P(fb + 3), P(fb + 1), 0, res, out, MTOK, D_MODEL, DFF_,
         DFF_, 0, 0, D_MODEL, 1, 0, 0, D_MODEL, 0, 0, 1.0f, 0, 1, 1);
  };

  const float* emb = P(174);

  // ----- encoder -----
  embed_k<<<MTOK, 256, 0, stream>>>(Cids, emb, P(273), cbuf);
  for (int l = 0; l < 6; ++l) {
    const int b = 175 + 16 * l;
    lnrm(cbuf, P(b + 13), P(b + 12), tbuf);        // ln1
    attn(tbuf, tbuf, b + 0, 0, cbuf, cbuf);        // self-attn + residual
    lnrm(cbuf, P(b + 15), P(b + 14), tbuf);        // ln2
    ffn(tbuf, b + 8, cbuf, cbuf);                  // ff + residual
  }
  lnrm(cbuf, P(272), P(271), cbuf);                // final encoder LN

  // ----- decoder -----
  embed_k<<<MTOK, 256, 0, stream>>>(Xids, emb, P(173), xbuf);
  for (int l = 0; l < 6; ++l) {
    const int b = 3 + 28 * l;
    lnrm(xbuf, P(b + 13), P(b + 12), tbuf);        // ln0
    attn(tbuf, tbuf, b + 20, 1, xbuf, xbuf);       // causal self-attn
    lnrm(xbuf, P(b + 15), P(b + 14), tbuf);        // ln1 (queries)
    lnrm(cbuf, P(b + 17), P(b + 16), t2buf);       // ln2 (enc K/V source)
    attn(tbuf, t2buf, b + 0, 0, xbuf, xbuf);       // cross-attn
    lnrm(xbuf, P(b + 19), P(b + 18), tbuf);        // ln3
    ffn(tbuf, b + 8, xbuf, xbuf);                  // ff + residual
  }
  lnrm(xbuf, P(172), P(171), tbuf);                // final decoder LN

  // ----- tied head: logits = x @ emb^T + head_b  (NT against emb[V][D]) -----
  gemm(tbuf, emb, P(274), 0, nullptr, (float*)d_out, MTOK, VOCAB, D_MODEL,
       D_MODEL, 0, 0, 1, D_MODEL, 0, 0, VOCAB, 0, 0, 1.0f, 0, 1, 1);
}
